// SeparableFCTP_29927332118764
// MI455X (gfx1250) — compile-verified, matrix operand order checked by reading, not run on hardware
//
#include <hip/hip_runtime.h>
#include <hip/hip_bf16.h>
#include <math.h>

typedef __attribute__((ext_vector_type(16))) _Float16 v16h;
typedef __attribute__((ext_vector_type(8)))  float    v8f;

#define NSC 128
#define NVC 64
#define FCC 64
#define WNC 448
#define INV_SQRT3 0.57735026918962576f
#define INV_SQRT2 0.70710678118654752f

#define OFF_W1F 0
#define OFF_W2F 4096
#define OFF_W3F 8192
#define OFF_WSF 36864
#define OFF_WVF 73728
#define OFF_WBUF 90112

__device__ __forceinline__ int kmap(int lane, int i) {
  return (i & 7) | (((i >> 3) & 1) << 4) | (((lane >> 4) & 1) << 3);
}
__device__ __forceinline__ float siluf(float x) { return x / (1.0f + __expf(-x)); }
__device__ __forceinline__ float sigmf(float x) { return 1.0f / (1.0f + __expf(-x)); }
__device__ __forceinline__ v8f vzero() {
  v8f z = {0.f, 0.f, 0.f, 0.f, 0.f, 0.f, 0.f, 0.f};
  return z;
}
__device__ __forceinline__ v8f wmma16(v16h a, v16h b, v8f c) {
  return __builtin_amdgcn_wmma_f32_16x16x32_f16(false, a, false, b, (short)0, c, false, false);
}

__device__ void packmat(const float* __restrict__ src, int K, int N,
                        _Float16* __restrict__ dst, int tid, int nth) {
  int total = K * N;
  int kcs = K >> 5;
  for (int idx = tid; idx < total; idx += nth) {
    int f = idx >> 9;
    int r = idx & 511;
    int l = r >> 4;
    int i = r & 15;
    int nt = f / kcs;
    int kc = f - nt * kcs;
    int n = nt * 16 + (l & 15);
    int k = kc * 32 + kmap(l, i);
    dst[idx] = (_Float16)src[k * N + n];
  }
}

__global__ void __launch_bounds__(256) kpack(
    const float* __restrict__ W1, const float* __restrict__ W2,
    const float* __restrict__ W3, const float* __restrict__ Ws,
    const float* __restrict__ Wv, _Float16* __restrict__ dst) {
  int tid = blockIdx.x * blockDim.x + threadIdx.x;
  int nth = gridDim.x * blockDim.x;
  packmat(W1, 64, 64, dst + OFF_W1F, tid, nth);
  packmat(W2, 64, 64, dst + OFF_W2F, tid, nth);
  packmat(W3, 64, 448, dst + OFF_W3F, tid, nth);
  packmat(Ws, 192, 192, dst + OFF_WSF, tid, nth);
  packmat(Wv, 256, 64, dst + OFF_WVF, tid, nth);
}

__global__ void __launch_bounds__(256) k1_mlp(
    const float* __restrict__ es, const _Float16* __restrict__ wpack,
    const float* __restrict__ b1, const float* __restrict__ b2,
    const float* __restrict__ offs, _Float16* __restrict__ wout,
    int E, int ntiles) {
  __shared__ __align__(32) _Float16 hb[8][1024];
  const int wave = threadIdx.x >> 5;
  const int lane = threadIdx.x & 31;
  int t = blockIdx.x * 8 + wave;
  if (t >= ntiles) t = ntiles - 1;
  const int row0 = t * 16;
  const _Float16* W1f = wpack + OFF_W1F;
  const _Float16* W2f = wpack + OFF_W2F;
  const _Float16* W3f = wpack + OFF_W3F;
  const int mhi = (lane >> 4) << 3;
  const int ncol = lane & 15;

  int rr = row0 + (lane & 15);
  if (rr >= E) rr = E - 1;

  v16h a0, a1;
#pragma unroll
  for (int i = 0; i < 16; ++i) {
    int k = kmap(lane, i);
    a0[i] = (_Float16)es[rr * FCC + k];
    a1[i] = (_Float16)es[rr * FCC + 32 + k];
  }

  // GEMM1 + silu -> hb (packed-A layout)
#pragma unroll
  for (int nt = 0; nt < 4; ++nt) {
    v8f acc = vzero();
    v16h bf = *(const v16h*)(W1f + (size_t)(nt * 2 + 0) * 512 + lane * 16);
    acc = wmma16(a0, bf, acc);
    bf = *(const v16h*)(W1f + (size_t)(nt * 2 + 1) * 512 + lane * 16);
    acc = wmma16(a1, bf, acc);
    int n = nt * 16 + ncol;
    float bias = b1[n];
#pragma unroll
    for (int i = 0; i < 8; ++i) {
      int m = i + mhi;
      float y = siluf(acc[i] + bias);
      int laneA = m + (((n >> 3) & 1) << 4);
      int iA = (n & 7) | (((n >> 4) & 1) << 3);
      hb[wave][((n >> 5) << 9) + laneA * 16 + iA] = (_Float16)y;
    }
  }
  __syncthreads();
  a0 = *(const v16h*)&hb[wave][lane * 16];
  a1 = *(const v16h*)&hb[wave][512 + lane * 16];
  __syncthreads();

  // GEMM2 + silu -> hb
#pragma unroll
  for (int nt = 0; nt < 4; ++nt) {
    v8f acc = vzero();
    v16h bf = *(const v16h*)(W2f + (size_t)(nt * 2 + 0) * 512 + lane * 16);
    acc = wmma16(a0, bf, acc);
    bf = *(const v16h*)(W2f + (size_t)(nt * 2 + 1) * 512 + lane * 16);
    acc = wmma16(a1, bf, acc);
    int n = nt * 16 + ncol;
    float bias = b2[n];
#pragma unroll
    for (int i = 0; i < 8; ++i) {
      int m = i + mhi;
      float y = siluf(acc[i] + bias);
      int laneA = m + (((n >> 3) & 1) << 4);
      int iA = (n & 7) | (((n >> 4) & 1) << 3);
      hb[wave][((n >> 5) << 9) + laneA * 16 + iA] = (_Float16)y;
    }
  }
  __syncthreads();
  a0 = *(const v16h*)&hb[wave][lane * 16];
  a1 = *(const v16h*)&hb[wave][512 + lane * 16];

  // GEMM3 + offset -> wout (f16)
  for (int nt = 0; nt < 28; ++nt) {
    v8f acc = vzero();
    v16h bf = *(const v16h*)(W3f + (size_t)(nt * 2 + 0) * 512 + lane * 16);
    acc = wmma16(a0, bf, acc);
    bf = *(const v16h*)(W3f + (size_t)(nt * 2 + 1) * 512 + lane * 16);
    acc = wmma16(a1, bf, acc);
    int n = nt * 16 + ncol;
    float bias = offs[n];
#pragma unroll
    for (int i = 0; i < 8; ++i) {
      int e = row0 + i + mhi;
      if (e < E) wout[(size_t)e * WNC + n] = (_Float16)(acc[i] + bias);
    }
  }
}

__global__ void __launch_bounds__(64) k2_tp(
    const float* __restrict__ node, const float* __restrict__ eattr,
    const _Float16* __restrict__ wpack, const _Float16* __restrict__ wbuf,
    const float* __restrict__ bs, const float* __restrict__ gs,
    const float* __restrict__ bn, const float* __restrict__ gv,
    float* __restrict__ out, int E, int ntiles) {
  __shared__ __align__(32) _Float16 st[2][15360];
  const int wave = threadIdx.x >> 5;
  const int lane = threadIdx.x & 31;
  int t = blockIdx.x * 2 + wave;
  if (t >= ntiles) t = ntiles - 1;
  const int row0 = t * 16;
  const _Float16* Wsf = wpack + OFF_WSF;
  const _Float16* Wvf = wpack + OFF_WVF;
  const int mhi = (lane >> 4) << 3;
  const int ncol = lane & 15;

  // build scal (16x192) in packed-A layout
  for (int idx = lane; idx < 3072; idx += 32) {
    int l = (idx >> 4) & 31;
    int i = idx & 15;
    int f = idx >> 9;
    int m = l & 15;
    int k = f * 32 + kmap(l, i);
    int e = row0 + m;
    int rr = (e < E) ? e : (E - 1);
    const float* ea = eattr + (size_t)rr * 4;
    const _Float16* wr = wbuf + (size_t)rr * WNC;
    float val;
    if (k < NSC) {
      val = (float)wr[k] * node[(size_t)rr * 320 + k] * ea[0];
    } else {
      int u = k - NSC;
      const float* vp = node + (size_t)rr * 320 + NSC + u * 3;
      float dot = vp[0] * ea[1] + vp[1] * ea[2] + vp[2] * ea[3];
      val = (float)wr[320 + u] * dot * INV_SQRT3;
    }
    st[wave][idx] = (_Float16)val;
  }
  // build vec_c (3 x 16x256) in packed-A layout
  for (int idx = lane; idx < 12288; idx += 32) {
    int c = idx >> 12;
    int r = idx & 4095;
    int l = (r >> 4) & 31;
    int i = r & 15;
    int f = r >> 9;
    int m = l & 15;
    int u = f * 32 + kmap(l, i);
    int e = row0 + m;
    int rr = (e < E) ? e : (E - 1);
    const float* ea = eattr + (size_t)rr * 4;
    const _Float16* wr = wbuf + (size_t)rr * WNC;
    float val;
    if (u < NSC) {
      val = (float)wr[NSC + u] * node[(size_t)rr * 320 + u] * ea[1 + c];
    } else if (u < NSC + NVC) {
      int q = u - NSC;
      val = (float)wr[256 + q] * node[(size_t)rr * 320 + NSC + q * 3 + c] * ea[0];
    } else {
      int q = u - NSC - NVC;
      const float* vp = node + (size_t)rr * 320 + NSC + q * 3;
      float cr;
      if (c == 0)      cr = vp[1] * ea[3] - vp[2] * ea[2];
      else if (c == 1) cr = vp[2] * ea[1] - vp[0] * ea[3];
      else             cr = vp[0] * ea[2] - vp[1] * ea[1];
      val = (float)wr[384 + q] * cr * INV_SQRT2;
    }
    st[wave][3072 + idx] = (_Float16)val;
  }
  __syncthreads();

  // s_lin = scal @ Ws
  v8f sacc[12];
#pragma unroll
  for (int nt = 0; nt < 12; ++nt) sacc[nt] = vzero();
#pragma unroll
  for (int kc = 0; kc < 6; ++kc) {
    v16h a = *(const v16h*)&st[wave][kc * 512 + lane * 16];
#pragma unroll
    for (int nt = 0; nt < 12; ++nt) {
      v16h b = *(const v16h*)(Wsf + (size_t)(nt * 6 + kc) * 512 + lane * 16);
      sacc[nt] = wmma16(a, b, sacc[nt]);
    }
  }

  // v_lin_c = vec_c @ Wv ; stash f32 results into freed LDS regions
  float* vlin0 = (float*)&st[wave][0];
  float* vlin1 = (float*)&st[wave][3072];
  float* vlin2 = (float*)&st[wave][7168];
  float* vl[3] = {vlin0, vlin1, vlin2};
#pragma unroll
  for (int c = 0; c < 3; ++c) {
    __syncthreads();
    v8f vacc[4];
#pragma unroll
    for (int nt = 0; nt < 4; ++nt) vacc[nt] = vzero();
#pragma unroll
    for (int kc = 0; kc < 8; ++kc) {
      v16h a = *(const v16h*)&st[wave][3072 + c * 4096 + kc * 512 + lane * 16];
#pragma unroll
      for (int nt = 0; nt < 4; ++nt) {
        v16h b = *(const v16h*)(Wvf + (size_t)(nt * 8 + kc) * 512 + lane * 16);
        vacc[nt] = wmma16(a, b, vacc[nt]);
      }
    }
#pragma unroll
    for (int nt = 0; nt < 4; ++nt) {
#pragma unroll
      for (int i = 0; i < 8; ++i) {
        int m = i + mhi;
        int o = nt * 16 + ncol;
        vl[c][m * 64 + o] = vacc[nt][i];
      }
    }
  }
  __syncthreads();

  // stage s_lin f32 (+ bias)
  float* slin = (float*)&st[wave][9216];
#pragma unroll
  for (int nt = 0; nt < 12; ++nt) {
    int n = nt * 16 + ncol;
    float bias = bs[n];
#pragma unroll
    for (int i = 0; i < 8; ++i) {
      int m = i + mhi;
      slin[m * 192 + n] = sacc[nt][i] + bias;
    }
  }
  __syncthreads();

  // norms + activations + output
  if (lane < 16) {
    int e = row0 + lane;
    if (e < E) {
      const float* sl = slin + lane * 192;
      float mean = 0.f;
      for (int j = 0; j < 192; ++j) mean += sl[j];
      mean *= (1.0f / 192.0f);
      float var = 0.f;
      for (int j = 0; j < 192; ++j) {
        float d = sl[j] - mean;
        var += d * d;
      }
      var *= (1.0f / 192.0f);
      float rs = rsqrtf(var + 1e-5f);
      float vs = 0.f;
      for (int o = 0; o < 64; ++o) {
        float x0 = vlin0[lane * 64 + o];
        float x1 = vlin1[lane * 64 + o];
        float x2 = vlin2[lane * 64 + o];
        vs += x0 * x0 + x1 * x1 + x2 * x2;
      }
      vs *= (1.0f / 64.0f);
      float rv = rsqrtf(vs + 1e-5f);
      float* orow = out + (size_t)e * 320;
      for (int j = 0; j < NSC; ++j) {
        float sn = gs[j] * (sl[j] - mean) * rs + bn[j];
        orow[j] = siluf(sn);
      }
      for (int q = 0; q < 64; ++q) {
        float sn = gs[NSC + q] * (sl[NSC + q] - mean) * rs + bn[NSC + q];
        float gate = sigmf(sn);
        float scale = rv * gate * gv[q];
        orow[NSC + q * 3 + 0] = vlin0[lane * 64 + q] * scale;
        orow[NSC + q * 3 + 1] = vlin1[lane * 64 + q] * scale;
        orow[NSC + q * 3 + 2] = vlin2[lane * 64 + q] * scale;
      }
    }
  }
}

extern "C" void kernel_launch(void* const* d_in, const int* in_sizes, int n_in,
                              void* d_out, int out_size, void* d_ws, size_t ws_size,
                              hipStream_t stream) {
  (void)n_in;
  (void)out_size;
  const float* node = (const float*)d_in[0];
  const float* eattr = (const float*)d_in[1];
  const float* es = (const float*)d_in[2];
  const float* W1 = (const float*)d_in[3];
  const float* b1 = (const float*)d_in[4];
  const float* W2 = (const float*)d_in[5];
  const float* b2 = (const float*)d_in[6];
  const float* W3 = (const float*)d_in[7];
  const float* offs = (const float*)d_in[8];
  const float* Ws = (const float*)d_in[9];
  const float* bs = (const float*)d_in[10];
  const float* Wv = (const float*)d_in[11];
  const float* gs = (const float*)d_in[12];
  const float* bn = (const float*)d_in[13];
  const float* gv = (const float*)d_in[14];

  int E = in_sizes[1] / 4;  // edge_attr is (E, 4)
  int ntiles = (E + 15) / 16;

  size_t need = (size_t)OFF_WBUF * 2 + (size_t)E * WNC * 2;
  if (ws_size < need) return;

  _Float16* wpack = (_Float16*)d_ws;
  _Float16* wbuf = wpack + OFF_WBUF;

  kpack<<<64, 256, 0, stream>>>(W1, W2, W3, Ws, Wv, wpack);
  k1_mlp<<<(ntiles + 7) / 8, 256, 0, stream>>>(es, wpack, b1, b2, offs, wbuf, E, ntiles);
  k2_tp<<<(ntiles + 1) / 2, 64, 0, stream>>>(node, eattr, wpack, wbuf, bs, gs, bn, gv,
                                             (float*)d_out, E, ntiles);
}
